// SpatioTemporalGCN_3882650436680
// MI455X (gfx1250) — matrix operand order repaired and checked by reading, not compiled, not run on hardware
//
#include <hip/hip_runtime.h>
#include <hip/hip_bf16.h>

// ---------------------------------------------------------------------------
// SpatioTemporalGCN for MI455X (gfx1250, wave32)
//   GEMMs           : bf16 WMMA (v_wmma_f32_16x16x32_bf16), f32 accumulate,
//                     A row-block staged to LDS via TDM tensor_load_to_lds
//   GCN scatter     : wave-per-edge float4 gather + global_atomic_add_f32
//   LSTM recurrence : persistent single-WGP kernel, 264KB of w_hh in LDS
// ---------------------------------------------------------------------------

typedef __attribute__((ext_vector_type(16))) __bf16 v16bf;
typedef __attribute__((ext_vector_type(8)))  float  v8f;
typedef unsigned int v4u  __attribute__((ext_vector_type(4)));
typedef int          v4i_ __attribute__((ext_vector_type(4)));
typedef int          v8i_ __attribute__((ext_vector_type(8)));

#define NODES   20000
#define EDGES   1280000
#define HDIM    128
#define GDIM    512      // 4*H gates
#define TOUT    12

// ---------------------------------------------------------------------------
// small utility kernels
// ---------------------------------------------------------------------------
__global__ void fill_kernel(float* p, float v, int n) {
    int i = blockIdx.x * blockDim.x + threadIdx.x;
    if (i < n) p[i] = v;
}

__global__ void deg_kernel(const long long* dst, float* deg, int e) {
    int i = blockIdx.x * blockDim.x + threadIdx.x;
    if (i < e) atomicAdd(&deg[(int)dst[i]], 1.0f);
}

__global__ void rsqrt_kernel(float* p, int n) {
    int i = blockIdx.x * blockDim.x + threadIdx.x;
    if (i < n) p[i] = rsqrtf(p[i]);
}

// ---------------------------------------------------------------------------
// WMMA GEMM:  C[M,N] = A[M,K] * B   (f32 memory, bf16 WMMA, f32 accum)
//   TRANSB=0 : B is [K,N] row-major        (layer weights)
//   TRANSB=1 : B is [N,K] row-major (B^T)  (w_ih)
// Block = 256 threads (8 waves) = one 16-row tile x 8 column tiles.
// A row-block (16xK f32) is DMA'd to LDS by the Tensor Data Mover.
// ---------------------------------------------------------------------------
template <int TRANSB, int K>
__global__ void __launch_bounds__(256)
gemm_wmma_bf16(const float* __restrict__ A,
               const float* __restrict__ B,
               const float* __restrict__ bias0,
               const float* __restrict__ bias1,
               float* __restrict__ C,
               int N, int ldb) {
    __shared__ float sA[16 * K];            // 8 KB for K=128

    const int wave = threadIdx.x >> 5;
    const int lane = threadIdx.x & 31;
    const int half = lane >> 4;             // 0: lanes 0-15, 1: lanes 16-31
    const int l16  = lane & 15;
    const int blocksPerRow = (N >> 4) >> 3; // column-tile groups of 8
    const int tm = blockIdx.x / blocksPerRow;
    const int tn = (blockIdx.x % blocksPerRow) * 8 + wave;
    const int n  = tn * 16 + l16;           // B/C column owned by this lane

#if __has_builtin(__builtin_amdgcn_tensor_load_to_lds) && \
    __has_builtin(__builtin_amdgcn_s_wait_tensorcnt)
    if (wave == 0) {
        // ---- Tensor DMA descriptor (ISA ch.8): 2-D tile 16 rows x K f32 ----
        unsigned long long ga =
            (unsigned long long)(uintptr_t)(A + (size_t)tm * 16 * K);
        unsigned ldsOff = (unsigned)(uintptr_t)(&sA[0]);   // low 32b = LDS offset
        v4u g0;
        g0[0] = 1u;                                        // count=1, user mode
        g0[1] = ldsOff;                                    // lds_addr
        g0[2] = (unsigned)(ga & 0xffffffffu);              // global_addr[31:0]
        g0[3] = (unsigned)((ga >> 32) & 0x01ffffffu)       // global_addr[56:32]
                | 0x80000000u;                             // type=2 (image)
        v8i_ g1;
        g1[0] = (int)(2u << 16);                           // data_size=4B
        g1[1] = (int)(((unsigned)K & 0xffffu) << 16);      // tensor_dim0 lo16
        g1[2] = (int)((((unsigned)K >> 16) & 0xffffu)      // tensor_dim0 hi16
                | (16u << 16));                            // tensor_dim1 lo16=16
        g1[3] = (int)((unsigned)K << 16);                  // tile_dim0=K (dim1 hi=0)
        g1[4] = 16;                                        // tile_dim1=16
        g1[5] = K;                                         // dim0_stride lo32 = K
        g1[6] = 0;                                         // dim0_stride hi / dim1_stride lo
        g1[7] = 0;
        v4i_ gz  = {0, 0, 0, 0};                           // 2-D: groups 2/3 unused
        v8i_ gz8 = {0, 0, 0, 0, 0, 0, 0, 0};               // unused extension group
        __builtin_amdgcn_tensor_load_to_lds(g0, g1, gz, gz, gz8, 0);
        __builtin_amdgcn_s_wait_tensorcnt(0);
    }
#else
    for (int idx = threadIdx.x; idx < 16 * K; idx += 256)
        sA[idx] = A[(size_t)tm * 16 * K + idx];
#endif
    __syncthreads();

    v8f acc = {};
#pragma unroll
    for (int k0 = 0; k0 < K; k0 += 32) {
        v16bf a, b;
        // A fragment from LDS: 16-bit A 16x32 layout (ISA 7.12.2):
        //   element p -> K = k0 + 16*(p>>3) + 8*half + (p&7); row = l16
#pragma unroll
        for (int p = 0; p < 16; ++p) {
            int ka = k0 + ((p >> 3) << 4) + (half << 3) + (p & 7);
            a[p] = (__bf16)sA[l16 * K + ka];
        }
        // B fragment: same K order as A: element p -> K = k0 + 16*(p>>3) + 8*half + (p&7)
#pragma unroll
        for (int p = 0; p < 16; ++p) {
            int kb = k0 + ((p >> 3) << 4) + (half << 3) + (p & 7);
            float bv = TRANSB ? B[n * ldb + kb] : B[kb * ldb + n];
            b[p] = (__bf16)bv;
        }
        acc = __builtin_amdgcn_wmma_f32_16x16x32_bf16(
            /*neg_a=*/false, a, /*neg_b=*/false, b,
            /*c_mod=*/(short)0, acc, /*reuse_a=*/false, /*reuse_b=*/false);
    }

    float bz = 0.0f;
    if (bias0) bz += bias0[n];
    if (bias1) bz += bias1[n];
    // C/D layout: VGPR r -> row = tm*16 + r + 8*half, col = n
#pragma unroll
    for (int r = 0; r < 8; ++r) {
        int row = tm * 16 + r + (half << 3);
        C[row * N + n] = acc[r] + bz;
    }
}

// ---------------------------------------------------------------------------
// GCN edge scatter: one wave per edge, 128 floats as float4 per lane,
// atomic f32 adds into the L2-resident accumulator.
// ---------------------------------------------------------------------------
__global__ void scatter_kernel(const long long* __restrict__ src,
                               const long long* __restrict__ dst,
                               const float* __restrict__ dinv,
                               const float* __restrict__ xw,
                               float* __restrict__ agg, int e) {
    const int edge = blockIdx.x * (blockDim.x >> 5) + (threadIdx.x >> 5);
    if (edge >= e) return;
    const int lane = threadIdx.x & 31;
    const int s = (int)src[edge];
    const int d = (int)dst[edge];
    const float norm = dinv[s] * dinv[d];
    const float4 v = ((const float4*)(xw + (size_t)s * HDIM))[lane];
    float* out = agg + (size_t)d * HDIM + lane * 4;
    atomicAdd(out + 0, v.x * norm);
    atomicAdd(out + 1, v.y * norm);
    atomicAdd(out + 2, v.z * norm);
    atomicAdd(out + 3, v.w * norm);
}

// self-loop contribution + bias + ReLU
__global__ void finish_layer_kernel(const float* __restrict__ agg,
                                    const float* __restrict__ xw,
                                    const float* __restrict__ dinv,
                                    const float* __restrict__ bias,
                                    float* __restrict__ out, int n_elem) {
    int i = blockIdx.x * blockDim.x + threadIdx.x;
    if (i >= n_elem) return;
    int row = i >> 7, col = i & (HDIM - 1);
    float di = dinv[row];
    float v = agg[i] + xw[i] * di * di + bias[col];
    out[i] = fmaxf(v, 0.0f);
}

// ---------------------------------------------------------------------------
// LSTM scan: strictly serial over 20000 steps. One persistent workgroup,
// 512 threads (one gate each). w_hh kept in LDS padded to 132 floats/row,
// h broadcast from LDS, c state in registers for threads 0..127.
// ---------------------------------------------------------------------------
#define WPAD 132
__global__ void __launch_bounds__(512)
lstm_scan_kernel(const float* __restrict__ xg,
                 const float* __restrict__ w_hh,
                 float* __restrict__ hs, int steps) {
    extern __shared__ float smem[];
    float* sW = smem;                       // GDIM * WPAD
    float* sH = sW + GDIM * WPAD;           // HDIM
    float* sG = sH + HDIM;                  // GDIM
    const int tid = threadIdx.x;            // 0..511

    for (int idx = tid; idx < GDIM * HDIM; idx += GDIM) {
        int r = idx >> 7, c = idx & (HDIM - 1);
        sW[r * WPAD + c] = w_hh[idx];
    }
    if (tid < HDIM) sH[tid] = 0.0f;
    float c_state = 0.0f;                   // meaningful for tid < 128
    const int gsel = tid >> 7;              // 0:i 1:f 2:g 3:o
    __syncthreads();

    const float4* wrow = (const float4*)(sW + tid * WPAD);
    const float4* hvec = (const float4*)sH;

    for (int t = 0; t < steps; ++t) {
        if (t + 8 < steps)                  // stream xg through L2 ahead of use
            __builtin_prefetch(&xg[(size_t)(t + 8) * GDIM + tid], 0, 0);
        float acc = xg[(size_t)t * GDIM + tid];
#pragma unroll 8
        for (int k4 = 0; k4 < HDIM / 4; ++k4) {
            float4 w = wrow[k4];
            float4 h = hvec[k4];
            acc = fmaf(w.x, h.x, acc);
            acc = fmaf(w.y, h.y, acc);
            acc = fmaf(w.z, h.z, acc);
            acc = fmaf(w.w, h.w, acc);
        }
        sG[tid] = (gsel == 2) ? tanhf(acc) : 1.0f / (1.0f + __expf(-acc));
        __syncthreads();
        if (tid < HDIM) {
            float ig = sG[tid];
            float fg = sG[tid + HDIM];
            float gg = sG[tid + 2 * HDIM];
            float og = sG[tid + 3 * HDIM];
            c_state = fg * c_state + ig * gg;
            float h = og * tanhf(c_state);
            sH[tid] = h;
            hs[(size_t)t * HDIM + tid] = h;
        }
        __syncthreads();
    }
}

// ---------------------------------------------------------------------------
// Final FC: out[n,t] = hs[n,:] . w_fc[t,:] + b_fc[t]; 16 nodes per block.
// ---------------------------------------------------------------------------
__global__ void fc_kernel(const float* __restrict__ hs,
                          const float* __restrict__ w_fc,
                          const float* __restrict__ b_fc,
                          float* __restrict__ out) {
    __shared__ float sh[16 * HDIM];
    __shared__ float sw[TOUT * HDIM];
    const int tid = threadIdx.x;            // 192 threads
    const int n0 = blockIdx.x * 16;
    for (int idx = tid; idx < 16 * HDIM; idx += 192)
        sh[idx] = hs[(size_t)n0 * HDIM + idx];
    for (int idx = tid; idx < TOUT * HDIM; idx += 192)
        sw[idx] = w_fc[idx];
    __syncthreads();
    const int ln = tid / TOUT;
    const int t  = tid % TOUT;
    float acc = b_fc[t];
#pragma unroll 4
    for (int k = 0; k < HDIM; ++k)
        acc = fmaf(sh[ln * HDIM + k], sw[t * HDIM + k], acc);
    out[(size_t)(n0 + ln) * TOUT + t] = acc;
}

// ---------------------------------------------------------------------------
extern "C" void kernel_launch(void* const* d_in, const int* in_sizes, int n_in,
                              void* d_out, int out_size, void* d_ws, size_t ws_size,
                              hipStream_t stream) {
    (void)in_sizes; (void)n_in; (void)out_size; (void)ws_size;
    const float*     x     = (const float*)d_in[0];
    const long long* ei    = (const long long*)d_in[1];   // int64 [2,E]
    const float*     W1    = (const float*)d_in[2];
    const float*     b1    = (const float*)d_in[3];
    const float*     W2    = (const float*)d_in[4];
    const float*     b2    = (const float*)d_in[5];
    const float*     w_ih  = (const float*)d_in[6];
    const float*     w_hh  = (const float*)d_in[7];
    const float*     b_ih  = (const float*)d_in[8];
    const float*     b_hh  = (const float*)d_in[9];
    const float*     w_fc  = (const float*)d_in[10];
    const float*     b_fc  = (const float*)d_in[11];
    float*           out   = (float*)d_out;

    // workspace layout (f32): dinv | xw | agg | hbuf | xg | hs
    float* dinv = (float*)d_ws;                   // N
    float* xw   = dinv + NODES;                   // N*H
    float* agg  = xw   + (size_t)NODES * HDIM;    // N*H
    float* hbuf = agg  + (size_t)NODES * HDIM;    // N*H
    float* xg   = hbuf + (size_t)NODES * HDIM;    // N*4H
    float* hs   = xg   + (size_t)NODES * GDIM;    // N*H

    const long long* src = ei;
    const long long* dst = ei + EDGES;

    const int nH = NODES * HDIM;
    const int rowTiles      = NODES / 16;              // 1250
    const int blocksLayer   = rowTiles * (HDIM / 128); // 1250 (8 col tiles/block)
    const int blocksXG      = rowTiles * (GDIM / 128); // 5000
    const int scatterBlocks = (EDGES + 7) / 8;         // 8 edges/block

    // degree -> dinv (self loop: start at 1.0)
    fill_kernel<<<(NODES + 255) / 256, 256, 0, stream>>>(dinv, 1.0f, NODES);
    deg_kernel<<<(EDGES + 255) / 256, 256, 0, stream>>>(dst, dinv, EDGES);
    rsqrt_kernel<<<(NODES + 255) / 256, 256, 0, stream>>>(dinv, NODES);

    // ---- GCN layer 1 ----
    gemm_wmma_bf16<0, HDIM><<<blocksLayer, 256, 0, stream>>>(
        x, W1, nullptr, nullptr, xw, HDIM, HDIM);
    fill_kernel<<<(nH + 255) / 256, 256, 0, stream>>>(agg, 0.0f, nH);
    scatter_kernel<<<scatterBlocks, 256, 0, stream>>>(src, dst, dinv, xw, agg, EDGES);
    finish_layer_kernel<<<(nH + 255) / 256, 256, 0, stream>>>(agg, xw, dinv, b1, hbuf, nH);

    // ---- GCN layer 2 ----
    gemm_wmma_bf16<0, HDIM><<<blocksLayer, 256, 0, stream>>>(
        hbuf, W2, nullptr, nullptr, xw, HDIM, HDIM);
    fill_kernel<<<(nH + 255) / 256, 256, 0, stream>>>(agg, 0.0f, nH);
    scatter_kernel<<<scatterBlocks, 256, 0, stream>>>(src, dst, dinv, xw, agg, EDGES);
    finish_layer_kernel<<<(nH + 255) / 256, 256, 0, stream>>>(agg, xw, dinv, b2, hbuf, nH);

    // ---- LSTM input projection: xg = h2 @ w_ih.T + (b_ih + b_hh) ----
    gemm_wmma_bf16<1, HDIM><<<blocksXG, 256, 0, stream>>>(
        hbuf, w_ih, b_ih, b_hh, xg, GDIM, HDIM);

    // ---- serial LSTM scan (single persistent workgroup) ----
    const size_t scanLds = (size_t)(GDIM * WPAD + HDIM + GDIM) * sizeof(float);
    lstm_scan_kernel<<<1, GDIM, scanLds, stream>>>(xg, w_hh, hs, NODES);

    // ---- final FC ----
    fc_kernel<<<NODES / 16, 192, 0, stream>>>(hs, w_fc, b_fc, out);
}
